// MyModel_87522843561086
// MI455X (gfx1250) — matrix operand order repaired and checked by reading, not compiled, not run on hardware
//
#include <hip/hip_runtime.h>
#include <hip/hip_bf16.h>
#include <cstdint>
#include <cstddef>

// ---------------------------------------------------------------------------
// 2-layer LSTM (H=100) + dense sigmoid head for MI455X (gfx1250, wave32).
//
//  * One persistent workgroup per 16 batch rows (B/16 = 32 WGs) loops over all
//    T timesteps; h/c state never leaves the WGP.
//  * z_t = [x_t | h_{t-1}] @ [W; U] + b as one WMMA K-chain per 16x16 tile
//    (V_WMMA_F32_16X16X32_F16, f32 accumulate). Input projection fused into
//    the recurrence -> no 839MB xw intermediate.
//  * Time-invariant B fragments (transposed [W;U;0], f16) preloaded into VGPRs
//    (uses the >256-VGPR CDNA5 wave32 file via VGPR-MSB addressing); the
//    steady-state inner loop is register-operand WMMAs only.
//  * Gate nonlinearities via v_exp_f32 + v_rcp_f32 (no IEEE div expansion);
//    two barriers per step (next-step input staging fused into gate phase).
//  * Layer-1 hidden sequence stored f16 (105MB in d_ws); layer 2 emits only
//    the final h (f32) for the tiny dense+sigmoid kernel.
// ---------------------------------------------------------------------------

typedef __attribute__((ext_vector_type(16))) _Float16 v16h;
typedef __attribute__((ext_vector_type(8)))  _Float16 v8h;
typedef __attribute__((ext_vector_type(8)))  float    v8f;

union V16 { v16h v; v8h h8[2]; };

__device__ __forceinline__ float fast_rcp(float x)  { return __builtin_amdgcn_rcpf(x); }
__device__ __forceinline__ float fast_sig(float x)  { return fast_rcp(1.0f + __expf(-x)); }
__device__ __forceinline__ float fast_tanh(float x) {
    return __builtin_fmaf(2.0f, fast_rcp(1.0f + __expf(-2.0f * x)), -1.0f);
}

// KIN  : input feature count (64 layer1, 100 layer2)
// KTOT : padded total K = pad32(KIN + H)  (192 / 224)
// INPUT_F16 : x is f32 (layer1), h1 is f16 (layer2)
// OUT_SEQ   : layer1 writes full f16 hidden sequence; layer2 only final h f32
template<int KIN, int KTOT, bool INPUT_F16, bool OUT_SEQ>
__launch_bounds__(256)
__global__ void lstm_layer_wmma(const void* __restrict__ xin,
                                const float* __restrict__ Wf,   // (KIN, 4H)
                                const float* __restrict__ Uf,   // (H,   4H)
                                const float* __restrict__ bf,   // (4H)
                                _Float16* __restrict__ hseq,    // (B,T,H) f16
                                float* __restrict__ hlast,      // (B,H)  f32
                                int T)
{
    constexpr int H    = 100;
    constexpr int G    = 400;              // 4*H gate width
    constexpr int NT   = G / 16;           // 25 N-tiles
    constexpr int KS   = KTOT / 32;        // WMMA K-steps per tile
    constexpr int NTH  = 256;              // 8 wave32s
    constexpr int NW   = NTH / 32;
    constexpr int MAXT = (NT + NW - 1) / NW;        // <=4 tiles per wave
    constexpr int ASTR = KTOT + 8;                  // padded A-stage row stride
    constexpr int ITX  = (KIN + 15) / 16;           // staging cols per thread
    constexpr int ITH  = (H + 15) / 16;             // gate cols per thread

    extern __shared__ __align__(16) char smem[];
    _Float16* BT   = (_Float16*)smem;                 // [G][KTOT]  transposed [W;U;0]
    _Float16* As   = BT + (size_t)G * KTOT;           // [16][ASTR] A-stage: [x_t | h | 0]
    float*    zb   = (float*)(As + 16 * ASTR);        // [16][G]    pre-activation z
    float*    cb   = zb + 16 * G;                     // [16][H]    cell state
    float*    bias = cb + 16 * H;                     // [G]

    const int tid   = threadIdx.x;
    const int bbase = blockIdx.x * 16;
    const int srow  = tid >> 4;    // staging/gate row 0..15
    const int scol  = tid & 15;    // staging/gate col base

    // ---- one-time: transpose+pack weights to LDS f16, zero h0/c0/pad,
    //      and stage x_0 (x region is NOT zero-filled: disjoint writes) ----
    for (int n = tid; n < G; n += NTH) bias[n] = bf[n];
    for (int idx = tid; idx < G * KTOT; idx += NTH) {
        int n = idx / KTOT, k = idx % KTOT;
        float v = 0.0f;
        if      (k < KIN)     v = Wf[(size_t)k * G + n];
        else if (k < KIN + H) v = Uf[(size_t)(k - KIN) * G + n];
        BT[idx] = (_Float16)v;
    }
    for (int idx = tid; idx < 16 * (KTOT - KIN); idx += NTH) {
        int m = idx / (KTOT - KIN), k = KIN + idx % (KTOT - KIN);
        As[m * ASTR + k] = (_Float16)0.0f;            // h0 = 0 and K-pad
    }
    #pragma unroll
    for (int it = 0; it < ITX; ++it) {
        const int k = scol + it * 16;
        if (k < KIN) {
            size_t g = ((size_t)(bbase + srow) * T + 0) * KIN + k;
            if constexpr (INPUT_F16) As[srow * ASTR + k] = ((const _Float16*)xin)[g];
            else                     As[srow * ASTR + k] = (_Float16)((const float*)xin)[g];
        }
    }
    for (int idx = tid; idx < 16 * H; idx += NTH) cb[idx] = 0.0f;
    __syncthreads();

    const int lane = tid & 31;
    const int wave = tid >> 5;
    const int m16  = lane & 15;   // row (A) or col (B/C) within the 16x16 tile
    const int kh   = lane >> 4;   // K-half select per ISA 16-bit WMMA layouts

    // ---- preload this wave's time-invariant B fragments + bias into VGPRs.
    // ISA 16-bit B 32x16 layout: lane holds column m16,
    // K = {32ks + 8kh + 0..7} and {32ks + 16 + 8kh + 0..7} (BT is K-contiguous).
    V16  bfrag[MAXT][KS];
    float bv[MAXT];
    #pragma unroll
    for (int tt = 0; tt < MAXT; ++tt) {
        const int nt = wave + tt * NW;
        if (nt < NT) {
            bv[tt] = bias[nt * 16 + m16];
            #pragma unroll
            for (int ks = 0; ks < KS; ++ks) {
                const _Float16* brow = &BT[(size_t)(nt * 16 + m16) * KTOT + ks * 32 + kh * 8];
                bfrag[tt][ks].h8[0] = *(const v8h*)brow;
                bfrag[tt][ks].h8[1] = *(const v8h*)(brow + 16);
            }
        }
    }
    __syncthreads();

    #pragma unroll 1
    for (int t = 0; t < T; ++t) {
        // ---- A fragments for this step (reused across this wave's N-tiles)
        // ISA 16-bit A 16x32 layout: lane holds row m16, halves
        // K = {32ks + 8kh + 0..7} and {32ks + 16 + 8kh + 0..7}.
        V16 afrag[KS];
        #pragma unroll
        for (int ks = 0; ks < KS; ++ks) {
            const int base = ks * 32 + kh * 8;
            afrag[ks].h8[0] = *(const v8h*)&As[m16 * ASTR + base];
            afrag[ks].h8[1] = *(const v8h*)&As[m16 * ASTR + base + 16];
        }

        // ---- z tile GEMM: register-resident B, K-chain of KS WMMAs ----
        #pragma unroll
        for (int tt = 0; tt < MAXT; ++tt) {
            const int nt = wave + tt * NW;
            if (nt < NT) {
                v8f acc;
                #pragma unroll
                for (int r = 0; r < 8; ++r) acc[r] = bv[tt];  // bias as acc init
                #pragma unroll
                for (int ks = 0; ks < KS; ++ks)
                    acc = __builtin_amdgcn_wmma_f32_16x16x32_f16(
                              false, afrag[ks].v, false, bfrag[tt][ks].v,
                              (short)0, acc, false, false);
                // C layout: lane -> n = m16, rows m = 8*kh + r
                float* zp = &zb[(8 * kh) * G + nt * 16 + m16];
                #pragma unroll
                for (int r = 0; r < 8; ++r) zp[r * G] = acc[r];
            }
        }
        __syncthreads();

        // ---- gates + state update (f32), write h back into A-stage,
        //      and stage NEXT step's x tile (disjoint As region) ----
        {
            const float* zr = &zb[srow * G];
            float*       cr = &cb[srow * H];
            #pragma unroll
            for (int it = 0; it < ITH; ++it) {
                const int j = scol + it * 16;
                if (j < H) {
                    float ig = fast_sig (zr[j]);
                    float fg = fast_sig (zr[H + j]);
                    float gg = fast_tanh(zr[2 * H + j]);
                    float og = fast_sig (zr[3 * H + j]);
                    float c  = fg * cr[j] + ig * gg;
                    cr[j] = c;
                    float h = og * fast_tanh(c);
                    As[srow * ASTR + KIN + j] = (_Float16)h;   // recurrent input t+1
                    if constexpr (OUT_SEQ)
                        hseq[((size_t)(bbase + srow) * T + t) * H + j] = (_Float16)h;
                    else if (t == T - 1)
                        hlast[(bbase + srow) * H + j] = h;
                }
            }
        }
        if (t + 1 < T) {
            #pragma unroll
            for (int it = 0; it < ITX; ++it) {
                const int k = scol + it * 16;
                if (k < KIN) {
                    size_t g = ((size_t)(bbase + srow) * T + (t + 1)) * KIN + k;
                    if constexpr (INPUT_F16) As[srow * ASTR + k] = ((const _Float16*)xin)[g];
                    else                     As[srow * ASTR + k] = (_Float16)((const float*)xin)[g];
                }
            }
        }
        // prefetch the step-after-next's rows (global_prefetch_b8)
        if (t + 2 < T && tid < 16) {
            size_t g = ((size_t)(bbase + tid) * T + (t + 2)) * (size_t)KIN;
            const char* p = INPUT_F16 ? (const char*)((const _Float16*)xin + g)
                                      : (const char*)((const float*)xin + g);
            __builtin_prefetch(p, 0, 1);
        }
        __syncthreads();
    }
}

// out[b] = sigmoid(hlast[b,:] . Wd + bd)
__global__ void dense_sigmoid_kernel(const float* __restrict__ hlast,
                                     const float* __restrict__ Wd,
                                     const float* __restrict__ bd,
                                     float* __restrict__ out, int B)
{
    int b = blockIdx.x * blockDim.x + threadIdx.x;
    if (b >= B) return;
    float acc = bd[0];
    const float* h = hlast + (size_t)b * 100;
    #pragma unroll 4
    for (int j = 0; j < 100; ++j) acc += h[j] * Wd[j];
    out[b] = fast_sig(acc);
}

extern "C" void kernel_launch(void* const* d_in, const int* in_sizes, int n_in,
                              void* d_out, int out_size, void* d_ws, size_t ws_size,
                              hipStream_t stream)
{
    constexpr int H = 100, G = 400, F = 64, T = 1024;
    const float* x  = (const float*)d_in[0];
    const float* W1 = (const float*)d_in[1];
    const float* U1 = (const float*)d_in[2];
    const float* b1 = (const float*)d_in[3];
    const float* W2 = (const float*)d_in[4];
    const float* U2 = (const float*)d_in[5];
    const float* b2 = (const float*)d_in[6];
    const float* Wd = (const float*)d_in[7];
    const float* bd = (const float*)d_in[8];
    float* out = (float*)d_out;

    const int B = in_sizes[0] / (T * F);   // 512

    // workspace: h1 sequence (f16) then final h2 (f32)
    _Float16* h1 = (_Float16*)d_ws;
    size_t seqBytes = ((size_t)B * T * H * sizeof(_Float16) + 255) & ~(size_t)255;
    float* h2last = (float*)((char*)d_ws + seqBytes);

    auto smemBytes = [](int KTOT) -> size_t {
        return (size_t)G * KTOT * sizeof(_Float16)          // BT
             + (size_t)16 * (KTOT + 8) * sizeof(_Float16)   // As (padded stride)
             + (size_t)16 * G * sizeof(float)               // zb
             + (size_t)16 * H * sizeof(float)               // cb
             + (size_t)G * sizeof(float);                   // bias
    };

    // Layer 1: x (f32, KIN=64, KTOT=192) -> h1 sequence (f16)
    lstm_layer_wmma<64, 192, false, true>
        <<<B / 16, 256, smemBytes(192), stream>>>(x, W1, U1, b1, h1, nullptr, T);

    // Layer 2: h1 (f16, KIN=100, KTOT=224) -> final h2 (f32)
    lstm_layer_wmma<100, 224, true, false>
        <<<B / 16, 256, smemBytes(224), stream>>>(h1, W2, U2, b2, nullptr, h2last, T);

    // Dense + sigmoid head
    dense_sigmoid_kernel<<<(B + 255) / 256, 256, 0, stream>>>(h2last, Wd, bd, out, B);
}
